// MultiHeadAttention_57397942944166
// MI455X (gfx1250) — matrix operand order repaired and checked by reading, not compiled, not run on hardware
//
#include <hip/hip_runtime.h>

#define B_  2
#define S_  2048
#define D_  1024
#define H_  16
#define DK_ 64
#define SROW 2052   // padded LDS row stride (floats) to break bank conflicts

typedef __attribute__((ext_vector_type(16))) __bf16 v16bf;
typedef __attribute__((ext_vector_type(8)))  __bf16 v8bf;
typedef __attribute__((ext_vector_type(8)))  float  v8f;

static __device__ __forceinline__ v16bf make_frag(v8bf lo, v8bf hi) {
  return __builtin_shufflevector(lo, hi, 0,1,2,3,4,5,6,7,8,9,10,11,12,13,14,15);
}

static __device__ __forceinline__ v8bf cvt8(float4 a, float4 b) {
  v8bf r;
  r[0]=(__bf16)a.x; r[1]=(__bf16)a.y; r[2]=(__bf16)a.z; r[3]=(__bf16)a.w;
  r[4]=(__bf16)b.x; r[5]=(__bf16)b.y; r[6]=(__bf16)b.z; r[7]=(__bf16)b.w;
  return r;
}

// A fragment (16x32, row-major MxK bf16 source).
// Lanes 0-15: M=lane, K in {0..7,16..23}; lanes 16-31: K in {8..15,24..31}.
static __device__ __forceinline__ v16bf load_a_bf16(const __bf16* base, int ld, int lane) {
  const int m   = lane & 15;
  const int kb0 = (lane < 16) ? 0 : 8;
  const __bf16* r = base + (long)m * ld + kb0;
  v8bf lo = *(const v8bf*)(r);
  v8bf hi = *(const v8bf*)(r + 16);
  return make_frag(lo, hi);
}

// A fragment from fp32 source (used for the LDS attn slab in phase 3).
static __device__ __forceinline__ v16bf load_a_f32(const float* base, int ld, int lane) {
  const int m   = lane & 15;
  const int kb0 = (lane < 16) ? 0 : 8;
  const float* r = base + (long)m * ld + kb0;
  float4 x0 = *(const float4*)(r);
  float4 x1 = *(const float4*)(r + 4);
  float4 y0 = *(const float4*)(r + 16);
  float4 y1 = *(const float4*)(r + 20);
  return make_frag(cvt8(x0, x1), cvt8(y0, y1));
}

// B fragment (32x16) with logical B[k][n] = Mrow[n][k] (multiply-by-transpose).
// Lane n=lane&15, K in {0..7,16..23} (lanes 0-15) or {8..15,24..31} (lanes 16-31).
static __device__ __forceinline__ v16bf load_bt_bf16(const __bf16* base, int ld, int lane) {
  const int n  = lane & 15;
  const int kb = (lane < 16) ? 0 : 8;
  const __bf16* r = base + (long)n * ld + kb;
  v8bf lo = *(const v8bf*)(r);
  v8bf hi = *(const v8bf*)(r + 16);
  return make_frag(lo, hi);
}

#define WMMA_BF16(a, b, c) \
  __builtin_amdgcn_wmma_f32_16x16x32_bf16(false, (a), false, (b), (short)0, (c), false, false)

// ---------------------------------------------------------------------------
// One-shot fp32 -> bf16 conversion (8 elements / thread, b128 in, b128 out).
// ---------------------------------------------------------------------------
__global__ __launch_bounds__(256) void mha_cvt_bf16_kernel(
    const float* __restrict__ in, __bf16* __restrict__ out, int n8)
{
  const int i = blockIdx.x * 256 + threadIdx.x;
  if (i >= n8) return;
  const float4* p = (const float4*)(in + (long)i * 8);
  float4 a = p[0], b = p[1];
  *(v8bf*)(out + (long)i * 8) = cvt8(a, b);
}

// ---------------------------------------------------------------------------
// GEMM core for the projections: 16x64 strip per wave, K pipelined with
// ping-pong fragment sets (no register rotation copies). D_ % 64 == 0.
// ---------------------------------------------------------------------------
static __device__ __forceinline__ void proj_gemm_core(
    const __bf16* __restrict__ X, const __bf16* __restrict__ W,
    int m0, int n0, int lane, v8f acc[4])
{
  v16bf aA = load_a_bf16(X + (long)m0 * D_, D_, lane);
  v16bf bA0 = load_bt_bf16(W + (long)(n0 +  0) * D_, D_, lane);
  v16bf bA1 = load_bt_bf16(W + (long)(n0 + 16) * D_, D_, lane);
  v16bf bA2 = load_bt_bf16(W + (long)(n0 + 32) * D_, D_, lane);
  v16bf bA3 = load_bt_bf16(W + (long)(n0 + 48) * D_, D_, lane);

  for (int k0 = 0; k0 < D_; k0 += 64) {
    // Load set B (k0+32) while computing with set A (k0).
    const int kb = k0 + 32;
    v16bf aB  = load_a_bf16(X + (long)m0 * D_ + kb, D_, lane);
    v16bf bB0 = load_bt_bf16(W + (long)(n0 +  0) * D_ + kb, D_, lane);
    v16bf bB1 = load_bt_bf16(W + (long)(n0 + 16) * D_ + kb, D_, lane);
    v16bf bB2 = load_bt_bf16(W + (long)(n0 + 32) * D_ + kb, D_, lane);
    v16bf bB3 = load_bt_bf16(W + (long)(n0 + 48) * D_ + kb, D_, lane);
    acc[0] = WMMA_BF16(aA, bA0, acc[0]);
    acc[1] = WMMA_BF16(aA, bA1, acc[1]);
    acc[2] = WMMA_BF16(aA, bA2, acc[2]);
    acc[3] = WMMA_BF16(aA, bA3, acc[3]);
    // Load set A (k0+64) while computing with set B (k0+32).
    const int ka = k0 + 64;
    if (ka < D_) {                    // uniform guard
      aA  = load_a_bf16(X + (long)m0 * D_ + ka, D_, lane);
      bA0 = load_bt_bf16(W + (long)(n0 +  0) * D_ + ka, D_, lane);
      bA1 = load_bt_bf16(W + (long)(n0 + 16) * D_ + ka, D_, lane);
      bA2 = load_bt_bf16(W + (long)(n0 + 32) * D_ + ka, D_, lane);
      bA3 = load_bt_bf16(W + (long)(n0 + 48) * D_ + ka, D_, lane);
    }
    acc[0] = WMMA_BF16(aB, bB0, acc[0]);
    acc[1] = WMMA_BF16(aB, bB1, acc[1]);
    acc[2] = WMMA_BF16(aB, bB2, acc[2]);
    acc[3] = WMMA_BF16(aB, bB3, acc[3]);
  }
}

// ---------------------------------------------------------------------------
// Projection: out = X @ W^T + bias. X bf16 (M=B*S x K=D), W bf16 (N=D x K=D).
// Result written bf16 in head-split layout:
//   layoutT==0 : out[((b*H+h)*S + s)*DK + dk]   (Q, K)
//   layoutT==1 : out[((b*H+h)*DK + dk)*S + s]   (V transposed for attn*V)
// ---------------------------------------------------------------------------
__global__ __launch_bounds__(256) void mha_proj_qkv_kernel(
    const __bf16* __restrict__ X, const __bf16* __restrict__ W,
    const float* __restrict__ bias, __bf16* __restrict__ out, int layoutT)
{
  const int lane  = threadIdx.x & 31;
  const int wave  = threadIdx.x >> 5;
  const int wg    = blockIdx.x * 8 + wave;   // 4096 waves total
  const int m0    = (wg >> 4) * 16;          // 256 row tiles
  const int n0    = (wg & 15) * 64;          // 16 column groups

  v8f acc[4] = {};
  proj_gemm_core(X, W, m0, n0, lane, acc);

  const int nl = lane & 15;
  const int mh = (lane < 16) ? 0 : 8;
#pragma unroll
  for (int nt = 0; nt < 4; ++nt) {
    const int n  = n0 + nt * 16 + nl;
    const float bb = bias[n];
    const int h  = n >> 6;      // n / DK_
    const int dk = n & 63;
#pragma unroll
    for (int r = 0; r < 8; ++r) {
      const int m  = m0 + mh + r;
      const int bi = m >> 11;   // m / S_
      const int s  = m & 2047;
      const float v = acc[nt][r] + bb;
      long idx;
      if (layoutT == 0) idx = ((long)(bi * H_ + h) * S_ + s) * DK_ + dk;
      else              idx = ((long)(bi * H_ + h) * DK_ + dk) * S_ + s;
      out[idx] = (__bf16)v;
    }
  }
}

// ---------------------------------------------------------------------------
// Fused attention per (b, h, 32 query rows): scores -> LDS (256KB slab),
// softmax in LDS, single fp32 attn write to HBM, then attn @ V from LDS.
// 512 threads = 16 waves.
// ---------------------------------------------------------------------------
__global__ __launch_bounds__(512) void mha_attn_fused_kernel(
    const __bf16* __restrict__ Q, const __bf16* __restrict__ K,
    const __bf16* __restrict__ VT, float* __restrict__ attn_out,
    __bf16* __restrict__ ctx_out)
{
  __shared__ float sc[32 * SROW];            // 32 x 2048 scores (padded rows)

  const int lane = threadIdx.x & 31;
  const int wave = threadIdx.x >> 5;         // 0..15
  const int q0   = blockIdx.x * 32;
  const int h    = blockIdx.y;
  const int b    = blockIdx.z;

  const __bf16* qh = Q  + ((long)(b * H_ + h) * S_)  * DK_;
  const __bf16* kh = K  + ((long)(b * H_ + h) * S_)  * DK_;
  const __bf16* vh = VT + ((long)(b * H_ + h) * DK_) * S_;

  // ---- Phase 1: scores = (Q K^T) * 1/sqrt(DK) into LDS ----
  // Hoist the Q fragments (2 q-tiles x 2 K-steps) into registers once.
  v16bf aQ[2][2];
#pragma unroll
  for (int mq = 0; mq < 2; ++mq)
#pragma unroll
    for (int kki = 0; kki < 2; ++kki)
      aQ[mq][kki] = load_a_bf16(qh + (long)(q0 + mq * 16) * DK_ + kki * 32, DK_, lane);

  for (int t = wave; t < 256; t += 16) {     // 2 q-tiles x 128 k-tiles
    const int mq = t & 1;
    const int nk = t >> 1;
    if (t + 16 < 256) {                      // pull next K-tile rows early
      const __bf16* pf = kh + (long)(((t + 16) >> 1) * 16) * DK_ + (lane & 15) * DK_;
      __builtin_prefetch(pf, 0, 1);          // global_prefetch_b8
    }
    v8f acc = {};
#pragma unroll
    for (int kki = 0; kki < 2; ++kki) {
      v16bf bf = load_bt_bf16(kh + (long)(nk * 16) * DK_ + kki * 32, DK_, lane);
      acc = WMMA_BF16(aQ[mq][kki], bf, acc);
    }
    const int n  = nk * 16 + (lane & 15);
    const int mh = (lane < 16) ? 0 : 8;
#pragma unroll
    for (int r = 0; r < 8; ++r)
      sc[(mq * 16 + mh + r) * SROW + n] = acc[r] * 0.125f;
  }
  __syncthreads();

  // ---- Phase 2: row softmax (2 rows per wave), write attn once ----
#pragma unroll
  for (int rr = 0; rr < 2; ++rr) {
    const int row = wave * 2 + rr;
    float* srow = sc + row * SROW;
    float mx = -1e30f;
    for (int c = lane; c < S_; c += 32) mx = fmaxf(mx, srow[c]);
#pragma unroll
    for (int off = 16; off > 0; off >>= 1) mx = fmaxf(mx, __shfl_xor(mx, off, 32));
    float sum = 0.f;
    for (int c = lane; c < S_; c += 32) {
      float e = __expf(srow[c] - mx);
      srow[c] = e;
      sum += e;
    }
#pragma unroll
    for (int off = 16; off > 0; off >>= 1) sum += __shfl_xor(sum, off, 32);
    const float inv = 1.0f / sum;
    float* grow = attn_out + ((long)(b * H_ + h) * S_ + (q0 + row)) * S_;
    for (int c = lane; c < S_; c += 32) {
      const float p = srow[c] * inv;
      srow[c] = p;
      grow[c] = p;      // the one mandatory HBM write of attn
    }
  }
  __syncthreads();

  // ---- Phase 3: context = attn @ V, split-K across wave pairs ----
  {
    const int t     = wave & 7;      // 2 q-tiles x 4 dk-tiles
    const int khalf = wave >> 3;
    const int mq    = t & 1;
    const int ndk   = t >> 1;
    v8f acc = {};
    const int ks = khalf * (S_ / 2);
    const __bf16* vrow = vh + (long)(ndk * 16) * S_;
    for (int kk = ks; kk < ks + S_ / 2; kk += 32) {
      if (kk + 256 < ks + S_ / 2)    // stream V^T one cacheline stride ahead
        __builtin_prefetch(vrow + (long)(lane & 15) * S_ + kk + 256, 0, 1);
      v16bf a  = load_a_f32(sc + (long)(mq * 16) * SROW + kk, SROW, lane); // LDS fp32 attn
      v16bf bf = load_bt_bf16(vrow + kk, S_, lane);                        // V^T contiguous
      acc = WMMA_BF16(a, bf, acc);
    }
    __syncthreads();                 // all LDS reads of sc are done
    if (khalf == 1) {                // upper-K waves park partials in (reused) LDS
#pragma unroll
      for (int r = 0; r < 8; ++r) sc[(t * 8 + r) * 32 + lane] = acc[r];
    }
    __syncthreads();
    if (khalf == 0) {
#pragma unroll
      for (int r = 0; r < 8; ++r) acc[r] += sc[(t * 8 + r) * 32 + lane];
      const int nl  = lane & 15;
      const int mh  = (lane < 16) ? 0 : 8;
      const int col = h * DK_ + ndk * 16 + nl;
#pragma unroll
      for (int r = 0; r < 8; ++r) {
        const int s = q0 + mq * 16 + mh + r;
        ctx_out[((long)b * S_ + s) * D_ + col] = (__bf16)acc[r];
      }
    }
  }
}

// ---------------------------------------------------------------------------
// Output projection: out = ctx(bf16) @ Wo^T(bf16) + bo, fp32 to d_out.
// ---------------------------------------------------------------------------
__global__ __launch_bounds__(256) void mha_proj_out_kernel(
    const __bf16* __restrict__ Xc, const __bf16* __restrict__ W,
    const float* __restrict__ bias, float* __restrict__ out)
{
  const int lane = threadIdx.x & 31;
  const int wave = threadIdx.x >> 5;
  const int wg   = blockIdx.x * 8 + wave;
  const int m0   = (wg >> 4) * 16;
  const int n0   = (wg & 15) * 64;

  v8f acc[4] = {};
  proj_gemm_core(Xc, W, m0, n0, lane, acc);

  const int nl = lane & 15;
  const int mh = (lane < 16) ? 0 : 8;
#pragma unroll
  for (int nt = 0; nt < 4; ++nt) {
    const int n = n0 + nt * 16 + nl;
    const float bb = bias[n];
#pragma unroll
    for (int r = 0; r < 8; ++r) {
      const int m = m0 + mh + r;
      out[(long)m * D_ + n] = acc[nt][r] + bb;
    }
  }
}

extern "C" void kernel_launch(void* const* d_in, const int* in_sizes, int n_in,
                              void* d_out, int out_size, void* d_ws, size_t ws_size,
                              hipStream_t stream) {
  (void)in_sizes; (void)n_in; (void)out_size; (void)ws_size;
  const float* query = (const float*)d_in[0];
  const float* key   = (const float*)d_in[1];
  const float* value = (const float*)d_in[2];
  const float* Wq = (const float*)d_in[3];
  const float* bq = (const float*)d_in[4];
  const float* Wk = (const float*)d_in[5];
  const float* bk = (const float*)d_in[6];
  const float* Wv = (const float*)d_in[7];
  const float* bv = (const float*)d_in[8];
  const float* Wo = (const float*)d_in[9];
  const float* bo = (const float*)d_in[10];

  float* out  = (float*)d_out;                       // (B,S,D)
  float* attn = out + (long)B_ * S_ * D_;            // (B,H,S,S)

  const long xElems = (long)B_ * S_ * D_;            // 4M elems
  const long wElems = (long)D_ * D_;                 // 1M elems
  __bf16* qb  = (__bf16*)d_ws;                       // (B,H,S,DK)
  __bf16* kb  = qb  + xElems;                        // (B,H,S,DK)
  __bf16* vtb = kb  + xElems;                        // (B,H,DK,S)
  __bf16* ctx = vtb + xElems;                        // (B*S, D)
  __bf16* xq  = ctx + xElems;                        // bf16 copies of inputs
  __bf16* xk  = xq  + xElems;
  __bf16* xv  = xk  + xElems;
  __bf16* wqb = xv  + xElems;                        // bf16 copies of weights
  __bf16* wkb = wqb + wElems;
  __bf16* wvb = wkb + wElems;
  __bf16* wob = wvb + wElems;

  // One-shot fp32 -> bf16 materialization (bandwidth-trivial pre-pass).
  const int xN8 = (int)(xElems / 8), wN8 = (int)(wElems / 8);
  mha_cvt_bf16_kernel<<<(xN8 + 255) / 256, 256, 0, stream>>>(query, xq,  xN8);
  mha_cvt_bf16_kernel<<<(xN8 + 255) / 256, 256, 0, stream>>>(key,   xk,  xN8);
  mha_cvt_bf16_kernel<<<(xN8 + 255) / 256, 256, 0, stream>>>(value, xv,  xN8);
  mha_cvt_bf16_kernel<<<(wN8 + 255) / 256, 256, 0, stream>>>(Wq,    wqb, wN8);
  mha_cvt_bf16_kernel<<<(wN8 + 255) / 256, 256, 0, stream>>>(Wk,    wkb, wN8);
  mha_cvt_bf16_kernel<<<(wN8 + 255) / 256, 256, 0, stream>>>(Wv,    wvb, wN8);
  mha_cvt_bf16_kernel<<<(wN8 + 255) / 256, 256, 0, stream>>>(Wo,    wob, wN8);

  // Q/K/V projections (bf16 in, bf16 head-split out); V pre-transposed.
  mha_proj_qkv_kernel<<<512, 256, 0, stream>>>(xq, wqb, bq, qb,  0);
  mha_proj_qkv_kernel<<<512, 256, 0, stream>>>(xk, wkb, bk, kb,  0);
  mha_proj_qkv_kernel<<<512, 256, 0, stream>>>(xv, wvb, bv, vtb, 1);

  // Fused scores + softmax + attn-write + context. Grid: (S/32, H, B).
  mha_attn_fused_kernel<<<dim3(S_ / 32, H_, B_), 512, 0, stream>>>(
      qb, kb, vtb, attn, ctx);

  // Output projection.
  mha_proj_out_kernel<<<512, 256, 0, stream>>>(ctx, wob, bo, out);
}
